// CirculantSSMLayer_62302795596497
// MI455X (gfx1250) — matrix operand order repaired and checked by reading, not compiled, not run on hardware
//
#include <hip/hip_runtime.h>
#include <hip/hip_bf16.h>
#include <math.h>

typedef __attribute__((ext_vector_type(16))) _Float16 v16h;
typedef __attribute__((ext_vector_type(8)))  _Float16 v8h;
typedef __attribute__((ext_vector_type(8)))  float    v8f;

#define BB 4
#define TT 4096
#define DM 1024
#define NS 256
#define MM (BB*TT)                 // 16384 rows
#define ANG_STEP 0.024543692606170259f   // 2*pi/256

// LDS tile geometry for the WMMA GEMM (rows padded: 32 -> 40 halves = 80B)
#define LPAD   40
#define ABYTES (128 * LPAD * 2)    // 10240 B per A buffer
#define BBYTES (64  * LPAD * 2)    // 5120 B per B buffer
#define SMEM_BYTES (2 * ABYTES + 2 * BBYTES)   // 30720 B

__device__ __forceinline__ float trig_angle(int prod) {
    return (float)(prod & (NS - 1)) * ANG_STEP;
}

__device__ __forceinline__ void async_load16(unsigned lds_off, const void* gaddr) {
    // CDNA5 async global -> LDS (ASYNCcnt); per-lane 16B
    asm volatile("global_load_async_to_lds_b128 %0, %1, off"
                 :: "v"(lds_off), "v"(gaddr) : "memory");
}

__device__ __forceinline__ void wait_async0() {
#if __has_builtin(__builtin_amdgcn_s_wait_asynccnt)
    __builtin_amdgcn_s_wait_asynccnt(0);
#else
    asm volatile("s_wait_asynccnt 0x0" ::: "memory");
#endif
}

// ---------------------------------------------------------------------------
// Prep kernels: fold DFT into weights, build f16 operands (cheap, per launch)
// ---------------------------------------------------------------------------

__global__ void convert_wa_kernel(const float* __restrict__ Wa, _Float16* __restrict__ wa) {
    int i = blockIdx.x * 256 + threadIdx.x;          // 256*1024 elems
    wa[i] = (_Float16)Wa[i];
}

// kb[n2][d]: n2<256 -> sum_s WB[s][d]*cos(2pi s n/256); n2>=256 -> -sum sin
__global__ void build_kb_kernel(const float* __restrict__ WB, _Float16* __restrict__ kb) {
    int gid = blockIdx.x * 256 + threadIdx.x;        // 512*1024 threads
    int d  = gid & (DM - 1);
    int n2 = gid >> 10;
    int n  = n2 & (NS - 1);
    bool im = n2 >= NS;
    float acc = 0.f;
    for (int s = 0; s < NS; ++s) {
        float ang = trig_angle(s * n);
        float t = im ? -__sinf(ang) : __cosf(ang);
        acc += WB[s * DM + d] * t;
    }
    kb[(size_t)n2 * DM + d] = (_Float16)acc;
}

__global__ void build_ubhat_kernel(const float* __restrict__ WBb, float* __restrict__ ub) {
    int n2 = blockIdx.x * 256 + threadIdx.x;         // 512 threads
    if (n2 >= 2 * NS) return;
    int n = n2 & (NS - 1);
    bool im = n2 >= NS;
    float acc = 0.f;
    for (int s = 0; s < NS; ++s) {
        float ang = trig_angle(s * n);
        acc += WBb[s] * (im ? -__sinf(ang) : __cosf(ang));
    }
    ub[n2] = acc;
}

__global__ void build_dft_kernel(_Float16* __restrict__ dft) {
    int gid = blockIdx.x * 256 + threadIdx.x;        // 512*256 threads
    int k  = gid & (NS - 1);
    int n2 = gid >> 8;
    int n  = n2 & (NS - 1);
    float ang = trig_angle(n * k);
    dft[gid] = (_Float16)((n2 < NS) ? __cosf(ang) : -__sinf(ang));
}

__global__ void build_g_kernel(const float* __restrict__ WC, _Float16* __restrict__ g) {
    int gid = blockIdx.x * 256 + threadIdx.x;        // 1024*512 threads
    int k2 = gid & (2 * NS - 1);
    int d  = gid >> 9;
    int n  = k2 & (NS - 1);
    bool im = k2 >= NS;
    float acc = 0.f;
    for (int k = 0; k < NS; ++k) {
        float ang = trig_angle(n * k);
        float t = im ? -__sinf(ang) : __cosf(ang);
        acc += WC[d * NS + k] * t;
    }
    g[(size_t)d * (2 * NS) + k2] = (_Float16)(acc * (1.f / NS));
}

// ---------------------------------------------------------------------------
// LayerNorm: one 256-thread block per token row; f16 output (WMMA A operand)
// ---------------------------------------------------------------------------
__global__ __launch_bounds__(256)
void ln_kernel(const float* __restrict__ x, const float* __restrict__ w,
               const float* __restrict__ b, _Float16* __restrict__ xn) {
    __shared__ float s1[256];
    __shared__ float s2[256];
    int row = blockIdx.x;
    int tid = threadIdx.x;
    const float4* xr = (const float4*)(x + (size_t)row * DM);
    float4 v = xr[tid];
    s1[tid] = v.x + v.y + v.z + v.w;
    s2[tid] = v.x * v.x + v.y * v.y + v.z * v.z + v.w * v.w;
    __syncthreads();
    for (int o = 128; o > 0; o >>= 1) {
        if (tid < o) { s1[tid] += s1[tid + o]; s2[tid] += s2[tid + o]; }
        __syncthreads();
    }
    float mu = s1[0] * (1.f / DM);
    float var = s2[0] * (1.f / DM) - mu * mu;
    float rs = rsqrtf(var + 1e-5f);
    int c = tid * 4;
    _Float16* o = xn + (size_t)row * DM + c;
    o[0] = (_Float16)((v.x - mu) * rs * w[c + 0] + b[c + 0]);
    o[1] = (_Float16)((v.y - mu) * rs * w[c + 1] + b[c + 1]);
    o[2] = (_Float16)((v.z - mu) * rs * w[c + 2] + b[c + 2]);
    o[3] = (_Float16)((v.w - mu) * rs * w[c + 3] + b[c + 3]);
}

// ---------------------------------------------------------------------------
// WMMA GEMM with async-LDS double-buffered staging.
// out[m][n] = epilogue( sum_k A[m][k] * B[n][k] ),  lda == ldb == K.
// Block tile 128x64, 8 waves (2x4), each wave does 4 stacked 16x16 tiles.
// Per K-step (32 halves): DMA next A/B slab to LDS (ASYNCcnt) while doing
// 4 WMMAs on the current slab read via ds loads.
// MODE 0: outH = f16( gamma * tanh(acc + bias[n]) )
// MODE 1: outF = acc (+ bias[n] if given)
// MODE 2: outF = acc + bias[n] + dskip[n]*xres[m][n]
// ---------------------------------------------------------------------------
template <int MODE>
__global__ __launch_bounds__(256)
void wmma_gemm_kernel(const _Float16* __restrict__ A,
                      const _Float16* __restrict__ Bm,
                      int K,
                      const float* __restrict__ bias,
                      const float* __restrict__ lg,
                      const float* __restrict__ xres,
                      const float* __restrict__ dskip,
                      float* __restrict__ outF,
                      _Float16* __restrict__ outH, int ldo) {
    __shared__ _Float16 smem[SMEM_BYTES / 2];

    const int tid  = threadIdx.x;
    const int lane = tid & 31;
    const int w    = tid >> 5;
    const int wm   = w >> 2;          // 0..1
    const int wn   = w & 3;           // 0..3
    const int hi   = lane >> 4;       // K-halves group
    const int lo16 = lane & 15;
    const int bm0 = blockIdx.y * 128;           // block row base
    const int bn0 = blockIdx.x * 64;            // block col base
    const int tn  = bn0 + wn * 16 + lo16;       // this lane's output column

    const unsigned smem_base = (unsigned)(uintptr_t)(&smem[0]);

    // ----- async DMA of one K-slab (32 halves) into LDS buffer `bsel` -----
    auto issue_slab = [&](int bsel, int k) {
        unsigned abase = smem_base + bsel * ABYTES;
        unsigned bbase = smem_base + 2 * ABYTES + bsel * BBYTES;
#pragma unroll
        for (int i = 0; i < 2; ++i) {           // A: 128 rows x 4 x 16B
            int cid = tid + i * 256;
            int row = cid >> 2, c = cid & 3;
            unsigned laddr = abase + row * (LPAD * 2) + c * 16;
            const void* g = (const void*)(A + (size_t)(bm0 + row) * K + k + c * 8);
            async_load16(laddr, g);
        }
        {                                       // B: 64 rows x 4 x 16B
            int row = tid >> 2, c = tid & 3;
            unsigned laddr = bbase + row * (LPAD * 2) + c * 16;
            const void* g = (const void*)(Bm + (size_t)(bn0 + row) * K + k + c * 8);
            async_load16(laddr, g);
        }
    };

    const int KT = K >> 5;
    issue_slab(0, 0);

    v8f acc[4] = {};
    for (int kt = 0; kt < KT; ++kt) {
        wait_async0();
        __syncthreads();
        if (kt + 1 < KT) issue_slab((kt + 1) & 1, (kt + 1) << 5);

        const _Float16* As = smem + ((kt & 1) * ABYTES) / 2;
        const _Float16* Bs = smem + (2 * ABYTES + (kt & 1) * BBYTES) / 2;

        const _Float16* bpl = Bs + (wn * 16 + lo16) * LPAD + hi * 8;
        v8h b0 = *(const v8h*)(bpl);
        v8h b1 = *(const v8h*)(bpl + 16);
        v16h bf = __builtin_shufflevector(b0, b1, 0, 1, 2, 3, 4, 5, 6, 7,
                                          8, 9, 10, 11, 12, 13, 14, 15);
#pragma unroll
        for (int r = 0; r < 4; ++r) {
            const _Float16* apl = As + (wm * 64 + r * 16 + lo16) * LPAD + hi * 8;
            v8h a0 = *(const v8h*)(apl);
            v8h a1 = *(const v8h*)(apl + 16);
            v16h af = __builtin_shufflevector(a0, a1, 0, 1, 2, 3, 4, 5, 6, 7,
                                              8, 9, 10, 11, 12, 13, 14, 15);
            acc[r] = __builtin_amdgcn_wmma_f32_16x16x32_f16(
                false, af, false, bf, (short)0, acc[r], false, false);
        }
    }

    float gamma = 0.f;
    if (MODE == 0) gamma = 1.f / (1.f + __expf(-lg[0]));

#pragma unroll
    for (int r = 0; r < 4; ++r) {
        int mbase = bm0 + wm * 64 + r * 16 + hi * 8;
#pragma unroll
        for (int e = 0; e < 8; ++e) {
            int m = mbase + e;
            float v = acc[r][e];
            if (MODE == 0) {
                v = gamma * tanhf(v + bias[tn]);
                outH[(size_t)m * ldo + tn] = (_Float16)v;
            } else if (MODE == 1) {
                if (bias) v += bias[tn];
                outF[(size_t)m * ldo + tn] = v;
            } else {
                v += bias[tn] + dskip[tn] * xres[(size_t)m * ldo + tn];
                outF[(size_t)m * ldo + tn] = v;
            }
        }
    }
}

// ---------------------------------------------------------------------------
// Magnitude-contracting scale on a_hat (couples re/im, post-DFT)
// ---------------------------------------------------------------------------
__global__ void scale_kernel(float* __restrict__ ah) {
    int idx = blockIdx.x * 256 + threadIdx.x;        // MM*NS threads
    int m = idx >> 8, n = idx & (NS - 1);
    size_t base = (size_t)m * (2 * NS);
    float re = ah[base + n];
    float im = ah[base + NS + n];
    float mag = fmaxf(sqrtf(re * re + im * im), 1e-8f);
    float s = 1.f / (1.f + __expf(-2.f * (1.f - mag)));
    ah[base + n]      = re * s;
    ah[base + NS + n] = im * s;
}

// ---------------------------------------------------------------------------
// Complex diagonal recurrence h_t = a_t*h_{t-1} + u_t over T, per (b, n).
// One wave32 per sequence; lane-local scan, Kogge-Stone shfl_up combine,
// then replay pass writing H = [Hr|Hi] f16.
// ---------------------------------------------------------------------------
__global__ __launch_bounds__(32)
void scan_kernel(const float* __restrict__ ah, const float* __restrict__ uh,
                 _Float16* __restrict__ H) {
    int seq = blockIdx.x;            // 0..1023
    int b = seq >> 8, n = seq & (NS - 1);
    int lane = threadIdx.x;          // 0..31 (wave32)
    const int CH = TT / 32;          // 128 timesteps per lane
    int t0 = lane * CH;

    float Ar = 1.f, Ai = 0.f, Br = 0.f, Bi = 0.f;
    for (int t = t0; t < t0 + CH; ++t) {
        size_t off = ((size_t)(b * TT + t)) * (2 * NS) + n;
        float ar = ah[off], ai = ah[off + NS];
        float ur = uh[off], ui = uh[off + NS];
        float nBr = ar * Br - ai * Bi + ur;
        float nBi = ar * Bi + ai * Br + ui;
        float nAr = Ar * ar - Ai * ai;
        float nAi = Ar * ai + Ai * ar;
        Br = nBr; Bi = nBi; Ar = nAr; Ai = nAi;
    }
    for (int o = 1; o < 32; o <<= 1) {
        float pAr = __shfl_up(Ar, o, 32);
        float pAi = __shfl_up(Ai, o, 32);
        float pBr = __shfl_up(Br, o, 32);
        float pBi = __shfl_up(Bi, o, 32);
        if (lane >= o) {
            float nBr = Ar * pBr - Ai * pBi + Br;
            float nBi = Ar * pBi + Ai * pBr + Bi;
            float nAr = pAr * Ar - pAi * Ai;
            float nAi = pAr * Ai + pAi * Ar;
            Br = nBr; Bi = nBi; Ar = nAr; Ai = nAi;
        }
    }
    float hr = __shfl_up(Br, 1, 32);
    float hi = __shfl_up(Bi, 1, 32);
    if (lane == 0) { hr = 0.f; hi = 0.f; }
    for (int t = t0; t < t0 + CH; ++t) {
        size_t off = ((size_t)(b * TT + t)) * (2 * NS) + n;
        float ar = ah[off], ai = ah[off + NS];
        float ur = uh[off], ui = uh[off + NS];
        float nr = ar * hr - ai * hi + ur;
        float ni = ar * hi + ai * hr + ui;
        hr = nr; hi = ni;
        H[off]      = (_Float16)hr;
        H[off + NS] = (_Float16)hi;
    }
}

// ---------------------------------------------------------------------------
extern "C" void kernel_launch(void* const* d_in, const int* in_sizes, int n_in,
                              void* d_out, int out_size, void* d_ws, size_t ws_size,
                              hipStream_t stream) {
    (void)in_sizes; (void)n_in; (void)out_size; (void)ws_size;
    const float* x    = (const float*)d_in[0];
    const float* Wa_w = (const float*)d_in[1];
    const float* Wa_b = (const float*)d_in[2];
    const float* lg   = (const float*)d_in[3];
    const float* WB_w = (const float*)d_in[4];
    const float* WB_b = (const float*)d_in[5];
    const float* WC_w = (const float*)d_in[6];
    const float* WC_b = (const float*)d_in[7];
    const float* Dsk  = (const float*)d_in[8];
    const float* ln_w = (const float*)d_in[9];
    const float* ln_b = (const float*)d_in[10];
    float* out = (float*)d_out;

    char* p = (char*)d_ws;
    auto carve = [&](size_t bytes) -> void* {
        void* r = (void*)p;
        p += (bytes + 255) & ~(size_t)255;
        return r;
    };
    _Float16* xn   = (_Float16*)carve((size_t)MM * DM * 2);        // 32 MB
    _Float16* wa   = (_Float16*)carve((size_t)NS * DM * 2);        // 0.5 MB
    _Float16* kb   = (_Float16*)carve((size_t)2 * NS * DM * 2);    // 1 MB
    float*    ubh  = (float*)   carve((size_t)2 * NS * 4);
    _Float16* dftm = (_Float16*)carve((size_t)2 * NS * NS * 2);    // 256 KB
    _Float16* g    = (_Float16*)carve((size_t)DM * 2 * NS * 2);    // 1 MB
    _Float16* af   = (_Float16*)carve((size_t)MM * NS * 2);        // 8 MB
    float*    ahat = (float*)   carve((size_t)MM * 2 * NS * 4);    // 32 MB
    float*    uhat = (float*)   carve((size_t)MM * 2 * NS * 4);    // 32 MB
    _Float16* H    = (_Float16*)carve((size_t)MM * 2 * NS * 2);    // 16 MB

    convert_wa_kernel<<<1024, 256, 0, stream>>>(Wa_w, wa);
    build_kb_kernel<<<2048, 256, 0, stream>>>(WB_w, kb);
    build_ubhat_kernel<<<2, 256, 0, stream>>>(WB_b, ubh);
    build_dft_kernel<<<512, 256, 0, stream>>>(dftm);
    build_g_kernel<<<2048, 256, 0, stream>>>(WC_w, g);

    ln_kernel<<<MM, 256, 0, stream>>>(x, ln_w, ln_b, xn);

    dim3 blk(256);
    // a = gamma * tanh(xn @ Wa^T + Wa_b)            [M x 256]
    wmma_gemm_kernel<0><<<dim3(NS / 64, MM / 128), blk, 0, stream>>>(
        xn, wa, DM, Wa_b, lg, nullptr, nullptr, nullptr, af, NS);
    // u_hat = xn @ (WB^T F) + F(WB_b)               [M x 512] = [re|im]
    wmma_gemm_kernel<1><<<dim3(2 * NS / 64, MM / 128), blk, 0, stream>>>(
        xn, kb, DM, ubh, nullptr, nullptr, nullptr, uhat, nullptr, 2 * NS);
    // a_hat(raw) = a @ F                            [M x 512]
    wmma_gemm_kernel<1><<<dim3(2 * NS / 64, MM / 128), blk, 0, stream>>>(
        af, dftm, NS, nullptr, nullptr, nullptr, nullptr, ahat, nullptr, 2 * NS);
    // a_hat *= sigmoid(2*(1-|a_hat|))
    scale_kernel<<<MM, 256, 0, stream>>>(ahat);
    // h_hat scan over time
    scan_kernel<<<1024, 32, 0, stream>>>(ahat, uhat, H);
    // y = [Hr|Hi] @ G + WC_b + D_skip * x           [M x 1024]
    wmma_gemm_kernel<2><<<dim3(DM / 64, MM / 128), blk, 0, stream>>>(
        H, g, 2 * NS, WC_b, nullptr, x, Dsk, out, nullptr, DM);
}
